// AttModel_65429531787917
// MI455X (gfx1250) — matrix operand order repaired and checked by reading, not compiled, not run on hardware
//
#include <hip/hip_runtime.h>
#include <hip/hip_bf16.h>

typedef __bf16 bf16_t;
typedef __attribute__((ext_vector_type(8)))  __bf16 v8bf;
typedef __attribute__((ext_vector_type(16))) __bf16 v16bf;
typedef __attribute__((ext_vector_type(8)))  float  v8f;

#define SHUF16(lo, hi) __builtin_shufflevector(lo, hi, 0,1,2,3,4,5,6,7,8,9,10,11,12,13,14,15)

// low 32 bits of a flat pointer into LDS == LDS byte offset (amdgcn flat->local truncation)
__device__ __forceinline__ unsigned lds_addr32(const void* p) {
  return (unsigned)(size_t)p;
}
__device__ __forceinline__ void async_ld_b128(unsigned lds_off, const void* g) {
  asm volatile("global_load_async_to_lds_b128 %0, %1, off"
               :: "v"(lds_off), "v"((unsigned long long)(size_t)g)
               : "memory");
}
__device__ __forceinline__ void wait_asynccnt0() {
  asm volatile("s_wait_asynccnt 0x0" ::: "memory");
}

// ---------------------------------------------------------------------------
// bf16 GEMM:  C[m][n] = act( sum_k A[m][k] * W[n][k] + bias[n] )   (= A @ W^T)
// Block = 4 waves = 64M x 64N tile. B k-slices (64 rows x 32 halves) staged
// into LDS with async global->LDS (double buffered, ASYNCcnt), rows padded to
// 40 halves (80B) for conflict-free ds_load_b128. Each wave: 1 M-tile x 4
// N-tiles (4 accumulators), A fragment reused 4x.
// Fragment k order: contiguous (k = 16*half + s), same for A and B (A 16x32 / B 32x16, wave32).
// ---------------------------------------------------------------------------
#define BROW 40   // halves per padded LDS row
__global__ void __launch_bounds__(128) k_gemm(
    const bf16_t* __restrict__ A, const bf16_t* __restrict__ W,
    const float* __restrict__ bias,
    float* __restrict__ Cf, bf16_t* __restrict__ Cb,
    int M, int Nn, int K, int act,
    const int* __restrict__ sel, int selt, int selclamp, long selstride)
{
  __shared__ __align__(16) bf16_t sB[2][64 * BROW];
  if (sel) {
    int s = sel[selt];
    if (selclamp && s < 0) s = 0;
    A += (long)s * selstride;
  }
  const int tilesM = M >> 4, tilesN = Nn >> 4;
  const int tngs = (tilesN + 3) >> 2;
  const int bidM = blockIdx.x / tngs;
  const int tng  = blockIdx.x - bidM * tngs;
  const int tid  = threadIdx.x;
  const int wv   = tid >> 5, lane = tid & 31;

  int tm = bidM * 4 + wv;
  const bool mvalid = (tm < tilesM);
  if (!mvalid) tm = tilesM - 1;

  const int half = lane >> 4;          // 0|1
  const int lo16 = lane & 15;
  const int kbA  = half * 16;          // A frag K base (contiguous 16)
  const int kbB  = half * 16;          // B frag K base (contiguous 16)
  const int n0   = tng * 64;
  const int nmax = Nn - 1;

  const bf16_t* arow = A + (long)(tm * 16 + lo16) * K;

  // stage one 64x32 B k-slice into sB[b]: 256 chunks of 16B, 2 per thread
  auto stage = [&](int kk, int b) {
#pragma unroll
    for (int i = 0; i < 2; ++i) {
      int c  = tid + (i << 7);         // 0..255
      int r  = c >> 2;                 // row 0..63
      int bo = (c & 3) << 4;           // 0,16,32,48 byte offset in row
      int n  = n0 + r; if (n > nmax) n = nmax;
      const char* g = (const char*)(W + (long)n * K + kk) + bo;
      unsigned loff = lds_addr32((const char*)&sB[b][0] + r * (BROW * 2) + bo);
      async_ld_b128(loff, g);
    }
  };

  v8f acc[4] = {v8f{}, v8f{}, v8f{}, v8f{}};
  stage(0, 0);

  int buf = 0;
  for (int kk = 0; kk < K; kk += 32, buf ^= 1) {
    wait_asynccnt0();
    __syncthreads();
    if (kk + 32 < K) {
      stage(kk + 32, buf ^ 1);
      __builtin_prefetch(arow + kk + 64, 0, 1);
    }
    v8bf a_lo = *(const v8bf*)(arow + kk + kbA);
    v8bf a_hi = *(const v8bf*)(arow + kk + kbA + 8);
    v16bf af = SHUF16(a_lo, a_hi);
    const bf16_t* sb = &sB[buf][0];
#pragma unroll
    for (int j = 0; j < 4; ++j) {
      const bf16_t* br = sb + (j * 16 + lo16) * BROW + kbB;
      v8bf b_lo = *(const v8bf*)br;
      v8bf b_hi = *(const v8bf*)(br + 8);
      v16bf bfrag = SHUF16(b_lo, b_hi);
      acc[j] = __builtin_amdgcn_wmma_f32_16x16x32_bf16(false, af, false, bfrag,
                                                       (short)0, acc[j], false, false);
    }
  }

  const int rowg = half * 8;
#pragma unroll
  for (int j = 0; j < 4; ++j) {
    int tn = tng * 4 + j;
    if (!mvalid || tn >= tilesN) continue;
    int col = tn * 16 + lo16;
    float bv = bias ? bias[col] : 0.0f;
#pragma unroll
    for (int i = 0; i < 8; ++i) {
      int row = tm * 16 + rowg + i;
      float v = acc[j][i] + bv;
      if (act) v = tanhf(v);
      long off = (long)row * Nn + col;
      if (Cf) Cf[off] = v;
      if (Cb) Cb[off] = (bf16_t)v;
    }
  }
}

// h = tanh(ha @ la_w^T + la_b + [use_f] hf @ lf_w^T + lf_b)   (M=32,N=512,K=512)
__global__ void __launch_bounds__(128) k_hcombine(
    const bf16_t* __restrict__ ha, const bf16_t* __restrict__ Wa, const float* __restrict__ ba,
    const bf16_t* __restrict__ hf, const bf16_t* __restrict__ Wf, const float* __restrict__ bfb,
    int use_f, float* __restrict__ Hf, bf16_t* __restrict__ Hb)
{
  int wid = blockIdx.x * 4 + (threadIdx.x >> 5);
  if (wid >= 64) return;       // 2 x 32 tiles
  const int lane = threadIdx.x & 31;
  const int tm = wid >> 5, tn = wid & 31;
  const int half = lane >> 4, lo16 = lane & 15;
  const int kbA = half * 16, kbB = half * 16;

  v8f acc = {};
  const bf16_t* arow = ha + (long)(tm * 16 + lo16) * 512;
  const bf16_t* brow = Wa + (long)(tn * 16 + lo16) * 512;
  for (int kk = 0; kk < 512; kk += 32) {
    v8bf a_lo = *(const v8bf*)(arow + kk + kbA);
    v8bf a_hi = *(const v8bf*)(arow + kk + kbA + 8);
    v8bf b_lo = *(const v8bf*)(brow + kk + kbB);
    v8bf b_hi = *(const v8bf*)(brow + kk + kbB + 8);
    v16bf af = SHUF16(a_lo, a_hi);
    v16bf bfrag = SHUF16(b_lo, b_hi);
    acc = __builtin_amdgcn_wmma_f32_16x16x32_bf16(false, af, false, bfrag,
                                                  (short)0, acc, false, false);
  }
  if (use_f) {
    arow = hf + (long)(tm * 16 + lo16) * 512;
    brow = Wf + (long)(tn * 16 + lo16) * 512;
    for (int kk = 0; kk < 512; kk += 32) {
      v8bf a_lo = *(const v8bf*)(arow + kk + kbA);
      v8bf a_hi = *(const v8bf*)(arow + kk + kbA + 8);
      v8bf b_lo = *(const v8bf*)(brow + kk + kbB);
      v8bf b_hi = *(const v8bf*)(brow + kk + kbB + 8);
      v16bf af = SHUF16(a_lo, a_hi);
      v16bf bfrag = SHUF16(b_lo, b_hi);
      acc = __builtin_amdgcn_wmma_f32_16x16x32_bf16(false, af, false, bfrag,
                                                    (short)0, acc, false, false);
    }
  }
  const int rowg = half * 8;
  const int col  = tn * 16 + lo16;
  float bv = ba[col] + bfb[col];
#pragma unroll
  for (int i = 0; i < 8; ++i) {
    int row = tm * 16 + rowg + i;
    float v = tanhf(acc[i] + bv);
    Hf[(long)row * 512 + col] = v;
    Hb[(long)row * 512 + col] = (bf16_t)v;
  }
}

// GRU elementwise. mode: 0=parent, 1=root(hprev=0), 2=brother(masked)
__global__ void k_gru_combine(const float* __restrict__ gi, const float* __restrict__ gh,
                              const float* __restrict__ hs_f, const int* __restrict__ sel,
                              int t, int mode, bf16_t* __restrict__ hout)
{
  int i = blockIdx.x * 256 + threadIdx.x;
  if (i >= 32 * 512) return;
  int b = i >> 9, j = i & 511;
  long gbase = (long)b * 1536;
  float ir = gi[gbase + j], iz = gi[gbase + 512 + j], in = gi[gbase + 1024 + j];
  float hr = gh[gbase + j], hz = gh[gbase + 512 + j], hn = gh[gbase + 1024 + j];
  float r = 1.0f / (1.0f + expf(-(ir + hr)));
  float z = 1.0f / (1.0f + expf(-(iz + hz)));
  float n = tanhf(in + r * hn);
  float hprev = 0.0f;
  bool zero_out = false;
  if (mode == 0) {
    int p = sel[t];
    hprev = hs_f[((long)p * 32 + b) * 512 + j];
  } else if (mode == 2) {
    int s = sel[t];
    if (s < 0) zero_out = true;
    else hprev = hs_f[((long)s * 32 + b) * 512 + j];
  }
  float h = (1.0f - z) * n + z * hprev;
  if (zero_out) h = 0.0f;
  hout[i] = (bf16_t)h;
}

// Attention: zeta=tanh(att_v+hg)@wh_w + wh_b, softmax over R=196, cont=alpha@att_e
__global__ void __launch_bounds__(256) k_attention(
    const float* __restrict__ att_v, const float* __restrict__ att_e,
    const float* __restrict__ hg, const float* __restrict__ wh_w,
    const float* __restrict__ wh_b, float* __restrict__ cont)
{
  __shared__ float s_hg[512];
  __shared__ float s_z[256];
  __shared__ float s_red[2];
  const int b = blockIdx.x;
  const int tid = threadIdx.x;
  for (int a = tid; a < 512; a += 256) s_hg[a] = hg[b * 512 + a];
  __syncthreads();
  const int wv = tid >> 5, lane = tid & 31;
  for (int r = wv; r < 196; r += 8) {
    const float* row = att_v + ((long)b * 196 + r) * 512;
    float s = 0.0f;
    for (int a = lane; a < 512; a += 32)
      s += tanhf(row[a] + s_hg[a]) * wh_w[a];
    for (int off = 16; off; off >>= 1) s += __shfl_xor(s, off, 32);
    if (lane == 0) s_z[r] = s + wh_b[0];
  }
  __syncthreads();
  if (tid < 32) {
    float mx = -1e30f;
    for (int r = tid; r < 196; r += 32) mx = fmaxf(mx, s_z[r]);
    for (int off = 16; off; off >>= 1) mx = fmaxf(mx, __shfl_xor(mx, off, 32));
    float sm = 0.0f;
    for (int r = tid; r < 196; r += 32) sm += expf(s_z[r] - mx);
    for (int off = 16; off; off >>= 1) sm += __shfl_xor(sm, off, 32);
    if (tid == 0) { s_red[0] = mx; s_red[1] = sm; }
  }
  __syncthreads();
  const float mx = s_red[0], inv = 1.0f / s_red[1];
  for (int r = tid; r < 196; r += 256) s_z[r] = expf(s_z[r] - mx) * inv;
  __syncthreads();
  for (int e = tid; e < 512; e += 256) {
    float acc = 0.0f;
    const float* ae = att_e + (long)b * 196 * 512 + e;
    for (int r = 0; r < 196; ++r) acc += s_z[r] * ae[(long)r * 512];
    cont[(long)b * 512 + e] = acc;
  }
}

__global__ void k_f2b(const float* __restrict__ s, bf16_t* __restrict__ d, long n)
{
  long i = (long)blockIdx.x * 256 + threadIdx.x;
  if (i < n) d[i] = (bf16_t)s[i];
}

// we_bf[t][b][e] = bf16(emb[word_idx[b][t]][e])   layout [N=121, B=32, E=512]
__global__ void k_gather_embed(const float* __restrict__ emb, const int* __restrict__ word_idx,
                               bf16_t* __restrict__ we)
{
  long i = (long)blockIdx.x * 256 + threadIdx.x;
  const long total = 121L * 32 * 512;
  if (i >= total) return;
  int e = (int)(i & 511);
  long bt = i >> 9;
  int t = (int)(bt / 32);
  int b = (int)(bt - (long)t * 32);
  int w = word_idx[b * 121 + t];
  we[i] = (bf16_t)emb[(long)w * 512 + e];
}

__global__ void k_fill_rows(float* __restrict__ dst, const float* __restrict__ bias,
                            int rows, int cols)
{
  long i = (long)blockIdx.x * 256 + threadIdx.x;
  if (i < (long)rows * cols) dst[i] = bias[i % cols];
}

// feats[(b*121+t)*1024 + c] : c<512 -> hs_bf[t][b][c] ; else bf16(conts[t][b][c-512])
__global__ void k_feats(const bf16_t* __restrict__ hs_bf, const float* __restrict__ conts,
                        bf16_t* __restrict__ feats)
{
  long i = (long)blockIdx.x * 256 + threadIdx.x;
  const long total = 3872L * 1024;
  if (i >= total) return;
  int c = (int)(i & 1023);
  long m = i >> 10;
  int b = (int)(m / 121);
  int t = (int)(m - (long)b * 121);
  if (c < 512) feats[i] = hs_bf[((long)t * 32 + b) * 512 + c];
  else         feats[i] = (bf16_t)conts[((long)t * 32 + b) * 512 + (c - 512)];
}

__global__ void k_target(const int* __restrict__ word_idx, int* __restrict__ out, int n)
{
  int i = blockIdx.x * 256 + threadIdx.x;
  if (i < n) out[i] = word_idx[i];
}

// ---------------------------------------------------------------------------
extern "C" void kernel_launch(void* const* d_in, const int* in_sizes, int n_in,
                              void* d_out, int out_size, void* d_ws, size_t ws_size,
                              hipStream_t stream)
{
  (void)in_sizes; (void)n_in; (void)out_size; (void)ws_size;
  const int B = 32, N = 121, R = 196, E = 512, H = 512, Aa = 512, V = 10000;
  const int Fc = 2048, Fa = 2048, G = 1536;

  const float* fc_feat   = (const float*)d_in[0];
  const float* att_feats = (const float*)d_in[1];
  const int*   word_idx  = (const int*)  d_in[2];
  const int*   parent    = (const int*)  d_in[3];
  const int*   brother   = (const int*)  d_in[4];
  const float* fc_w  = (const float*)d_in[5];  const float* fc_b  = (const float*)d_in[6];
  const float* ae_w  = (const float*)d_in[7];  const float* ae_b  = (const float*)d_in[8];
  const float* wv_w  = (const float*)d_in[9];  const float* wv_b  = (const float*)d_in[10];
  const float* wg_w  = (const float*)d_in[11]; const float* wg_b  = (const float*)d_in[12];
  const float* wh_w  = (const float*)d_in[13]; const float* wh_b  = (const float*)d_in[14];
  const float* emb   = (const float*)d_in[15];
  const float* a_wih = (const float*)d_in[16]; const float* a_whh = (const float*)d_in[17];
  const float* a_bih = (const float*)d_in[18]; const float* a_bhh = (const float*)d_in[19];
  const float* f_wih = (const float*)d_in[20]; const float* f_whh = (const float*)d_in[21];
  const float* f_bih = (const float*)d_in[22]; const float* f_bhh = (const float*)d_in[23];
  const float* la_w  = (const float*)d_in[24]; const float* la_b  = (const float*)d_in[25];
  const float* lf_w  = (const float*)d_in[26]; const float* lf_b  = (const float*)d_in[27];
  const float* lo_w  = (const float*)d_in[28]; const float* lo_b  = (const float*)d_in[29];

  // -------- workspace carve-out --------
  char* wp = (char*)d_ws;
  auto alloc = [&](size_t bytes) -> void* {
    void* r = (void*)wp;
    wp += (bytes + 255) & ~(size_t)255;
    return r;
  };
  bf16_t* ae_w_bf   = (bf16_t*)alloc(sizeof(bf16_t) * (size_t)E * Fa);
  bf16_t* wv_w_bf   = (bf16_t*)alloc(sizeof(bf16_t) * (size_t)Aa * E);
  bf16_t* wg_w_bf   = (bf16_t*)alloc(sizeof(bf16_t) * (size_t)Aa * H);
  bf16_t* fc_w_bf   = (bf16_t*)alloc(sizeof(bf16_t) * (size_t)E * Fc);
  bf16_t* fc_ft_bf  = (bf16_t*)alloc(sizeof(bf16_t) * (size_t)B * Fc);
  bf16_t* att_ft_bf = (bf16_t*)alloc(sizeof(bf16_t) * (size_t)B * R * Fa);
  bf16_t* a_wih_bf  = (bf16_t*)alloc(sizeof(bf16_t) * (size_t)G * E);
  bf16_t* a_whh_bf  = (bf16_t*)alloc(sizeof(bf16_t) * (size_t)G * H);
  bf16_t* f_wih_bf  = (bf16_t*)alloc(sizeof(bf16_t) * (size_t)G * E);
  bf16_t* f_whh_bf  = (bf16_t*)alloc(sizeof(bf16_t) * (size_t)G * H);
  bf16_t* la_w_bf   = (bf16_t*)alloc(sizeof(bf16_t) * (size_t)H * H);
  bf16_t* lf_w_bf   = (bf16_t*)alloc(sizeof(bf16_t) * (size_t)H * H);
  bf16_t* lo_w_bf   = (bf16_t*)alloc(sizeof(bf16_t) * (size_t)V * (H + E));
  bf16_t* att_e_bf  = (bf16_t*)alloc(sizeof(bf16_t) * (size_t)B * R * E);
  bf16_t* fc_e_bf   = (bf16_t*)alloc(sizeof(bf16_t) * (size_t)B * E);
  bf16_t* we_bf     = (bf16_t*)alloc(sizeof(bf16_t) * (size_t)N * B * E);
  bf16_t* hs_bf     = (bf16_t*)alloc(sizeof(bf16_t) * (size_t)N * B * H);
  bf16_t* ha_bf     = (bf16_t*)alloc(sizeof(bf16_t) * (size_t)B * H);
  bf16_t* hf_bf     = (bf16_t*)alloc(sizeof(bf16_t) * (size_t)B * H);
  bf16_t* feats_bf  = (bf16_t*)alloc(sizeof(bf16_t) * (size_t)B * N * (H + E));
  float*  att_e_f   = (float*)alloc(sizeof(float) * (size_t)B * R * E);
  float*  att_v_f   = (float*)alloc(sizeof(float) * (size_t)B * R * Aa);
  float*  gi_a      = (float*)alloc(sizeof(float) * (size_t)B * G);
  float*  gh_a      = (float*)alloc(sizeof(float) * (size_t)B * G);
  float*  gi_f      = (float*)alloc(sizeof(float) * (size_t)B * G);
  float*  gh_f      = (float*)alloc(sizeof(float) * (size_t)B * G);
  float*  hs_f      = (float*)alloc(sizeof(float) * (size_t)N * B * H);
  float*  conts     = (float*)alloc(sizeof(float) * (size_t)N * B * E);
  float*  hg_f      = (float*)alloc(sizeof(float) * (size_t)B * Aa);

  auto cvt = [&](const float* s, bf16_t* d, long n) {
    k_f2b<<<(unsigned)((n + 255) / 256), 256, 0, stream>>>(s, d, n);
  };
  auto gemm = [&](const bf16_t* A, const bf16_t* W, const float* bias,
                  float* Cf, bf16_t* Cb, int M, int Nn, int K, int act,
                  const int* sel, int selt, int selclamp, long selstride) {
    int tmgs = (M / 16 + 3) / 4;
    int tngs = (Nn / 16 + 3) / 4;
    k_gemm<<<tmgs * tngs, 128, 0, stream>>>(A, W, bias, Cf, Cb, M, Nn, K, act,
                                            sel, selt, selclamp, selstride);
  };

  // -------- precompute: bf16 weight/activation copies --------
  cvt(ae_w,  ae_w_bf,  (long)E * Fa);
  cvt(wv_w,  wv_w_bf,  (long)Aa * E);
  cvt(wg_w,  wg_w_bf,  (long)Aa * H);
  cvt(fc_w,  fc_w_bf,  (long)E * Fc);
  cvt(fc_feat, fc_ft_bf, (long)B * Fc);
  cvt(att_feats, att_ft_bf, (long)B * R * Fa);
  cvt(a_wih, a_wih_bf, (long)G * E);
  cvt(a_whh, a_whh_bf, (long)G * H);
  cvt(f_wih, f_wih_bf, (long)G * E);
  cvt(f_whh, f_whh_bf, (long)G * H);
  cvt(la_w,  la_w_bf,  (long)H * H);
  cvt(lf_w,  lf_w_bf,  (long)H * H);
  cvt(lo_w,  lo_w_bf,  (long)V * (H + E));
  k_gather_embed<<<(N * B * E + 255) / 256, 256, 0, stream>>>(emb, word_idx, we_bf);

  // -------- big precompute GEMMs --------
  gemm(att_ft_bf, ae_w_bf, ae_b, att_e_f, att_e_bf, B * R, E, Fa, 0, nullptr, 0, 0, 0);
  gemm(att_e_bf,  wv_w_bf, wv_b, att_v_f, nullptr,  B * R, Aa, E, 0, nullptr, 0, 0, 0);
  gemm(fc_ft_bf,  fc_w_bf, fc_b, nullptr, fc_e_bf,  B, E, Fc, 0, nullptr, 0, 0, 0);

  // -------- root node (t = 0) --------
  gemm(fc_e_bf, a_wih_bf, a_bih, gi_a, nullptr, B, G, E, 0, nullptr, 0, 0, 0);
  k_fill_rows<<<(B * G + 255) / 256, 256, 0, stream>>>(gh_a, a_bhh, B, G);
  k_gru_combine<<<64, 256, 0, stream>>>(gi_a, gh_a, hs_f, parent, 0, 1, ha_bf);
  k_hcombine<<<16, 128, 0, stream>>>(ha_bf, la_w_bf, la_b, ha_bf, lf_w_bf, lf_b,
                                     0, hs_f, hs_bf);
  gemm(hs_bf, wg_w_bf, wg_b, hg_f, nullptr, B, Aa, H, 0, nullptr, 0, 0, 0);
  k_attention<<<B, 256, 0, stream>>>(att_v_f, att_e_f, hg_f, wh_w, wh_b, conts);

  // -------- sequential tree recurrence --------
  const long strideBE = (long)B * E;   // == B*H
  for (int t = 1; t < N; ++t) {
    gemm(we_bf, a_wih_bf, a_bih, gi_a, nullptr, B, G, E, 0, parent,  t, 0, strideBE);
    gemm(hs_bf, a_whh_bf, a_bhh, gh_a, nullptr, B, G, H, 0, parent,  t, 0, strideBE);
    gemm(we_bf, f_wih_bf, f_bih, gi_f, nullptr, B, G, E, 0, brother, t, 1, strideBE);
    gemm(hs_bf, f_whh_bf, f_bhh, gh_f, nullptr, B, G, H, 0, brother, t, 1, strideBE);
    k_gru_combine<<<64, 256, 0, stream>>>(gi_a, gh_a, hs_f, parent,  t, 0, ha_bf);
    k_gru_combine<<<64, 256, 0, stream>>>(gi_f, gh_f, hs_f, brother, t, 2, hf_bf);
    k_hcombine<<<16, 128, 0, stream>>>(ha_bf, la_w_bf, la_b, hf_bf, lf_w_bf, lf_b, 1,
                                       hs_f + (long)t * B * H, hs_bf + (long)t * B * H);
    gemm(hs_bf + (long)t * B * H, wg_w_bf, wg_b, hg_f, nullptr, B, Aa, H, 0,
         nullptr, 0, 0, 0);
    k_attention<<<B, 256, 0, stream>>>(att_v_f, att_e_f, hg_f, wh_w, wh_b,
                                       conts + (long)t * B * E);
  }

  // -------- output head --------
  k_feats<<<(int)((3872L * 1024 + 255) / 256), 256, 0, stream>>>(hs_bf, conts, feats_bf);
  gemm(feats_bf, lo_w_bf, lo_b, (float*)d_out, nullptr, B * N, V, H + E, 0,
       nullptr, 0, 0, 0);
  k_target<<<(B * N + 255) / 256, 256, 0, stream>>>(
      word_idx, (int*)((float*)d_out + (long)B * N * V), B * N);
}
